// NormalizedHungarianLoss_64003602645338
// MI455X (gfx1250) — compile-verified, matrix operand order checked by reading, not run on hardware
//
#include <hip/hip_runtime.h>

// ---------------------------------------------------------------------------
// NormalizedHungarianLoss for MI455X (gfx1250, wave32)
//   Kernel 1: one workgroup per batch runs Jonker-Volgenant LAP with the
//             column scan parallelized across 512 threads; LAP state in LDS.
//   Kernel 2: one wave32 reduces the 32x512 matched values with
//             V_WMMA_F32_16X16X4_F32 (B = ones => accumulator holds sums).
// ---------------------------------------------------------------------------

constexpr int kB = 32;    // batches
constexpr int kN = 512;   // matrix side
constexpr int kNP1 = kN + 1;

typedef float    v2f  __attribute__((ext_vector_type(2)));
typedef float    v8f  __attribute__((ext_vector_type(8)));
typedef _Float16 v16h __attribute__((ext_vector_type(16)));

__global__ __launch_bounds__(kN)
void hungarian_lap_kernel(const float* __restrict__ D, float* __restrict__ ws) {
  const int b = blockIdx.x;
  const int t = threadIdx.x;        // 0..511
  const int j = t + 1;              // this thread owns column j (1..512)
  const float* __restrict__ Db = D + (size_t)b * kN * kN;

  __shared__ float u_[kNP1], v_[kNP1], minv_[kNP1];
  __shared__ int   p_[kNP1], way_[kNP1], used_[kNP1];
  __shared__ float red_val[kN];
  __shared__ int   red_idx[kN];
  __shared__ float sc_scale, sc_shift;

  const float INF = __builtin_inff();

  // ---- per-batch min / max for the (affine) normalization --------------
  float lmin = INF, lmax = -INF;
  for (int k = t; k < kN * kN; k += kN) {
    float x = Db[k];
    lmin = fminf(lmin, x);
    lmax = fmaxf(lmax, x);
  }
  red_val[t] = lmin; __syncthreads();
  for (int s = kN / 2; s > 0; s >>= 1) {
    if (t < s) red_val[t] = fminf(red_val[t], red_val[t + s]);
    __syncthreads();
  }
  const float dmin = red_val[0];
  __syncthreads();
  red_val[t] = lmax; __syncthreads();
  for (int s = kN / 2; s > 0; s >>= 1) {
    if (t < s) red_val[t] = fmaxf(red_val[t], red_val[t + s]);
    __syncthreads();
  }
  if (t == 0) {
    float rng = red_val[0] - dmin + 1e-8f;
    sc_scale = 1.0f / rng;
    sc_shift = -dmin / rng;
  }

  // ---- LAP state init ---------------------------------------------------
  u_[j] = 0.0f; v_[j] = 0.0f; p_[j] = 0;
  if (t == 0) { u_[0] = 0.0f; v_[0] = 0.0f; p_[0] = 0; }
  __syncthreads();
  const float scale = sc_scale;
  const float shift = sc_shift;

  // ---- Jonker-Volgenant over rows --------------------------------------
  for (int i = 1; i <= kN; ++i) {
    minv_[j] = INF; used_[j] = 0;
    if (t == 0) { p_[0] = i; used_[0] = 0; minv_[0] = INF; }
    __syncthreads();

    int j0 = 0;
    for (;;) {
      if (t == 0) used_[j0] = 1;
      __syncthreads();

      const int   i0  = p_[j0];
      const float ui0 = u_[i0];

      float key;
      if (!used_[j]) {
        // normalized cost, fused affine: (D - dmin)/range == D*scale + shift
        float cij = Db[(size_t)(i0 - 1) * kN + (j - 1)] * scale + shift;
        float cur = cij - ui0 - v_[j];
        if (cur < minv_[j]) { minv_[j] = cur; way_[j] = j0; }
        key = minv_[j];
      } else {
        key = INF;
      }

      // block argmin (lowest index wins ties, matching np.argmin)
      red_val[t] = key; red_idx[t] = j;
      __syncthreads();
      for (int s = kN / 2; s > 0; s >>= 1) {
        if (t < s) {
          float ov = red_val[t + s];
          int   oi = red_idx[t + s];
          if (ov < red_val[t] || (ov == red_val[t] && oi < red_idx[t])) {
            red_val[t] = ov; red_idx[t] = oi;
          }
        }
        __syncthreads();
      }
      const float delta = red_val[0];
      const int   j1    = red_idx[0];

      // dual / slack updates
      if (used_[j]) {
        u_[p_[j]] += delta;       // p injective over used columns: no race
        v_[j]     -= delta;
      } else {
        minv_[j]  -= delta;
      }
      if (t == 0) { u_[p_[0]] += delta; v_[0] -= delta; }  // column 0 always used
      __syncthreads();

      j0 = j1;
      if (p_[j0] == 0) break;     // free column reached -> augment
    }

    // augment alternating path (cheap, serial on thread 0)
    if (t == 0) {
      int jj = j0;
      while (jj != 0) {
        int jn = way_[jj];
        p_[jj] = p_[jn];
        jj = jn;
      }
    }
    __syncthreads();
  }

  // ---- emit matched values of the ORIGINAL D ---------------------------
  // column j is matched to row p_[j]-1; order within the batch is irrelevant
  // for the final mean.
  {
    int r = p_[j] - 1;
    ws[(size_t)b * kN + t] = Db[(size_t)r * kN + t];   // D[b][r][j-1], t==j-1
  }
}

// ---------------------------------------------------------------------------
// Final mean over kB*kN matched values using WMMA (B = ones => row sums).
// Single wave32, EXEC all ones, no divergence before the WMMA ops.
// ---------------------------------------------------------------------------
__global__ __launch_bounds__(32)
void matched_mean_wmma_kernel(const float* __restrict__ ws, float* __restrict__ out) {
  const int lane = threadIdx.x;   // 0..31
  v8f c = {};

#if __has_builtin(__builtin_amdgcn_wmma_f32_16x16x4_f32)
  // A: 16x4 f32 tile = 64 values per step (any bijection of 64 values into
  // the tile works since B is all-ones). D[m][n] accumulates rowsum_m(A);
  // total sum = column 0 of D = lane0 + lane16 per-lane VGPR sums.
  v2f bm; bm[0] = 1.0f; bm[1] = 1.0f;
  for (int it = 0; it < (kB * kN) / 64; ++it) {
    v2f a;
    a[0] = ws[it * 64 + 2 * lane];
    a[1] = ws[it * 64 + 2 * lane + 1];
    c = __builtin_amdgcn_wmma_f32_16x16x4_f32(
        /*neg_a=*/false, a, /*neg_b=*/false, bm,
        /*c_mod=*/(short)0, c, /*reuse_a=*/false, /*reuse_b=*/false);
  }
#else
  // Fallback: codegen-confirmed f16 WMMA; 512 values per step.
  v16h bm;
  for (int k = 0; k < 16; ++k) bm[k] = (_Float16)1.0f;
  for (int it = 0; it < (kB * kN) / 512; ++it) {
    v16h a;
    for (int k = 0; k < 16; ++k) a[k] = (_Float16)ws[it * 512 + lane * 16 + k];
    c = __builtin_amdgcn_wmma_f32_16x16x32_f16(
        false, a, false, bm, (short)0, c, false, false);
  }
#endif

  // C/D layout: lanes 0-15 hold N=lane (M=0..7 in VGPR 0..7),
  //             lanes 16-31 hold N=lane-16 (M=8..15).
  // Sum of column N=0 = s(lane 0) + s(lane 16) = total sum.
  float s = 0.0f;
  for (int k = 0; k < 8; ++k) s += c[k];
  float s0  = __shfl(s, 0, 32);
  float s16 = __shfl(s, 16, 32);
  if (lane == 0) out[0] = (s0 + s16) * (1.0f / (float)(kB * kN));
}

extern "C" void kernel_launch(void* const* d_in, const int* in_sizes, int n_in,
                              void* d_out, int out_size, void* d_ws, size_t ws_size,
                              hipStream_t stream) {
  (void)in_sizes; (void)n_in; (void)out_size; (void)ws_size;
  const float* D  = (const float*)d_in[0];      // [32, 512, 512] f32
  float*       out = (float*)d_out;             // scalar f32
  float*       ws  = (float*)d_ws;              // needs kB*kN floats = 64 KB

  hungarian_lap_kernel<<<kB, kN, 0, stream>>>(D, ws);
  matched_mean_wmma_kernel<<<1, 32, 0, stream>>>(ws, out);
}
